// MultiHeadSelfAttention_88691074662718
// MI455X (gfx1250) — matrix operand order repaired and checked by reading, not compiled, not run on hardware
//
#include <hip/hip_runtime.h>

typedef __bf16 bf16_t;
typedef __attribute__((ext_vector_type(16))) __bf16 v16bf;
typedef __attribute__((ext_vector_type(8)))  float  v8f;
typedef int vint4_t __attribute__((vector_size(16)));

#define DIM  1024
#define SEQ  2048
#define NB   2
#define NH   16
#define DH   64
#define ROWS (NB * SEQ)   // 4096

// ---------------- gfx1250 async memory->LDS path (guarded, with fallback) ----
#if defined(__has_builtin)
#if __has_builtin(__builtin_amdgcn_global_load_async_to_lds_b128) && \
    __has_builtin(__builtin_amdgcn_s_wait_asynccnt)
#define USE_ASYNC_LDS 1
#endif
#endif
#ifndef USE_ASYNC_LDS
#define USE_ASYNC_LDS 0
#endif

__device__ __forceinline__ void stage_b128(const bf16_t* g, bf16_t* l) {
#if USE_ASYNC_LDS
    __builtin_amdgcn_global_load_async_to_lds_b128(
        (__attribute__((address_space(1))) vint4_t*)g,
        (__attribute__((address_space(3))) vint4_t*)l, 0, 0);
#else
    *(uint4*)l = *(const uint4*)g;
#endif
}

__device__ __forceinline__ void async_stage_wait() {
#if USE_ASYNC_LDS
    __builtin_amdgcn_s_wait_asynccnt(0);
#endif
}

// ---------------------------------------------------------------- RMSNorm
__launch_bounds__(256)
__global__ void rmsnorm_kernel(const float* __restrict__ x,
                               const float* __restrict__ scale,
                               bf16_t* __restrict__ xn) {
    int row = blockIdx.x;
    int tid = threadIdx.x;
    const float4* xr = (const float4*)(x + (size_t)row * DIM);
    float4 v = xr[tid];
    float ss = v.x * v.x + v.y * v.y + v.z * v.z + v.w * v.w;
    #pragma unroll
    for (int off = 16; off > 0; off >>= 1) ss += __shfl_xor(ss, off, 32);
    __shared__ float wsum[8];
    if ((tid & 31) == 0) wsum[tid >> 5] = ss;
    __syncthreads();
    float tot = 0.f;
    #pragma unroll
    for (int i = 0; i < 8; ++i) tot += wsum[i];
    float inv = rsqrtf(tot / (float)DIM + 1e-6f);
    const float4* sr = (const float4*)scale;
    float4 s4 = sr[tid];
    bf16_t* o = xn + (size_t)row * DIM + tid * 4;
    o[0] = (bf16_t)(v.x * inv * s4.x);
    o[1] = (bf16_t)(v.y * inv * s4.y);
    o[2] = (bf16_t)(v.z * inv * s4.z);
    o[3] = (bf16_t)(v.w * inv * s4.w);
}

// ---------------------------------------------------------------- fp32 -> bf16
__launch_bounds__(256)
__global__ void f32_to_bf16_kernel(const float* __restrict__ in,
                                   bf16_t* __restrict__ out, int n) {
    int i = (blockIdx.x * blockDim.x + threadIdx.x) * 4;
    if (i + 3 < n) {
        float4 v = *(const float4*)(in + i);
        out[i + 0] = (bf16_t)v.x;
        out[i + 1] = (bf16_t)v.y;
        out[i + 2] = (bf16_t)v.z;
        out[i + 3] = (bf16_t)v.w;
    }
}

// ---------------------------------------------------------------- RoPE (fp32 in, bf16 out)
__launch_bounds__(256)
__global__ void rope_kernel(const float* __restrict__ in,
                            bf16_t* __restrict__ out) {
    int idx = blockIdx.x * blockDim.x + threadIdx.x;  // ROWS * NH * 32 threads
    int row = idx >> 9;
    int rem = idx & 511;
    int h = rem >> 5, i = rem & 31;
    int pos = row & (SEQ - 1);
    const float ln_base = 13.815510557964274f;  // ln(1e6)
    float theta = __expf(-(float)i * (ln_base / 32.0f));
    float angle = (float)pos * theta;
    float s, c;
    __sincosf(angle, &s, &c);
    size_t base = (size_t)row * DIM + h * DH + i;
    float x1 = in[base], x2 = in[base + 32];
    out[base]      = (bf16_t)(x1 * c - x2 * s);
    out[base + 32] = (bf16_t)(x1 * s + x2 * c);
}

// ---------------------------------------------------------------- bf16 WMMA GEMM
// C[M,N] (f32) = A[M,K] (bf16, row-major) * B[K,N] (bf16, row-major)
// 128x128 block tile, 8 waves; each wave: 2 m-tiles x 4 n-tiles = 8 WMMAs / K-step.
#define BM 128
#define BN 128
#define BK 32
#define LDA_S 40
#define LDB_S 40

__launch_bounds__(256)
__global__ void gemm_bf16_kernel(const bf16_t* __restrict__ A,
                                 const bf16_t* __restrict__ B,
                                 float* __restrict__ C,
                                 int M, int N, int K) {
    __shared__ bf16_t As[BM * LDA_S];   // [m][k]
    __shared__ bf16_t Bt[BN * LDB_S];   // [n][k]  (transposed at staging)
    int tid = threadIdx.x;
    int lane = tid & 31, w = tid >> 5;
    int half = lane >> 4, l15 = lane & 15;
    int m0 = blockIdx.y * BM, n0 = blockIdx.x * BN;
    int mo = (w & 3) * 32;              // wave's 2 m-tiles: mo, mo+16
    int no = (w >> 2) * 64;             // wave's 4 n-tiles: no + 16*t
    v8f acc[2][4];
    #pragma unroll
    for (int i = 0; i < 2; ++i)
        #pragma unroll
        for (int t = 0; t < 4; ++t) acc[i][t] = (v8f){};

    int arow = tid >> 1, acol = (tid & 1) * 16;   // A: 128 rows, 2 thr/row, 16 elems each
    int brow = tid >> 3, bcol = (tid & 7) * 16;   // B: 32 rows, 8 thr/row, 16 elems each

    for (int k0 = 0; k0 < K; k0 += BK) {
        // stage A 128x32 (straight copy -> async-to-LDS when available)
        const bf16_t* ag = A + (size_t)(m0 + arow) * K + k0 + acol;
        stage_b128(ag,     As + arow * LDA_S + acol);
        stage_b128(ag + 8, As + arow * LDA_S + acol + 8);
        // stage B 32x128, transposed into Bt[n][k]
        const bf16_t* bg = B + (size_t)(k0 + brow) * N + n0 + bcol;
        uint4 bv0 = *(const uint4*)(bg);
        uint4 bv1 = *(const uint4*)(bg + 8);
        const bf16_t* bh0 = (const bf16_t*)&bv0;
        const bf16_t* bh1 = (const bf16_t*)&bv1;
        #pragma unroll
        for (int j = 0; j < 8; ++j) Bt[(bcol + j) * LDB_S + brow]     = bh0[j];
        #pragma unroll
        for (int j = 0; j < 8; ++j) Bt[(bcol + 8 + j) * LDB_S + brow] = bh1[j];
        if (k0 + BK < K) {
            __builtin_prefetch(ag + BK, 0, 3);
            __builtin_prefetch(B + (size_t)(k0 + BK + brow) * N + n0 + bcol, 0, 3);
        }
        async_stage_wait();
        __syncthreads();

        // A fragments (2 m-tiles): lane l15 = row; K chunks at half*16, half*16+8
        union { uint4 u[2]; v16bf v; } af[2];
        int kb = half * 16;
        #pragma unroll
        for (int i = 0; i < 2; ++i) {
            const bf16_t* ap = As + (mo + i * 16 + l15) * LDA_S;
            af[i].u[0] = *(const uint4*)(ap + kb);
            af[i].u[1] = *(const uint4*)(ap + kb + 8);
        }
        // B fragments (4 n-tiles), each reused by 2 WMMAs
        int ks = half * 16;
        #pragma unroll
        for (int t = 0; t < 4; ++t) {
            union { uint4 u[2]; v16bf v; } bf;
            const bf16_t* bp = Bt + (no + t * 16 + l15) * LDB_S;
            bf.u[0] = *(const uint4*)(bp + ks);
            bf.u[1] = *(const uint4*)(bp + ks + 8);
            acc[0][t] = __builtin_amdgcn_wmma_f32_16x16x32_bf16(
                false, af[0].v, false, bf.v, (short)0, acc[0][t], false, false);
            acc[1][t] = __builtin_amdgcn_wmma_f32_16x16x32_bf16(
                false, af[1].v, false, bf.v, (short)0, acc[1][t], false, false);
        }
        __syncthreads();
    }
    #pragma unroll
    for (int i = 0; i < 2; ++i)
        #pragma unroll
        for (int t = 0; t < 4; ++t)
            #pragma unroll
            for (int r = 0; r < 8; ++r) {
                int row = m0 + mo + i * 16 + r + 8 * half;
                C[(size_t)row * N + n0 + no + t * 16 + l15] = acc[i][t][r];
            }
}

// ---------------------------------------------------------------- flash attention
#define QB 128
#define KB 32
#define KS_S 72
#define VT_S 40
#define PS_S 40

__launch_bounds__(256)
__global__ void flash_attn_kernel(const bf16_t* __restrict__ Q,
                                  const bf16_t* __restrict__ Km,
                                  const bf16_t* __restrict__ V,
                                  bf16_t* __restrict__ O) {
    __shared__ bf16_t Ks[KB * KS_S];        // [key][d]
    __shared__ bf16_t Vt[DH * VT_S];        // [d][key]
    __shared__ bf16_t Ps[8 * 16 * PS_S];    // per-wave P scratch [row][key]
    int tid = threadIdx.x, lane = tid & 31, w = tid >> 5;
    int half = lane >> 4, l15 = lane & 15;
    int q0 = blockIdx.x * QB;
    int bh = blockIdx.y;
    int bidx = bh >> 4, h = bh & 15;
    size_t rowbase = (size_t)bidx * SEQ;
    int qw0 = q0 + w * 16;

    // preload Q fragments for dh split into two k=32 steps
    union { uint4 u[2]; v16bf v; } qf0, qf1;
    {
        size_t qrow = (rowbase + qw0 + l15) * DIM + h * DH;
        int kb = half * 16;
        qf0.u[0] = *(const uint4*)(Q + qrow + kb);
        qf0.u[1] = *(const uint4*)(Q + qrow + kb + 8);
        qf1.u[0] = *(const uint4*)(Q + qrow + 32 + kb);
        qf1.u[1] = *(const uint4*)(Q + qrow + 32 + kb + 8);
    }

    v8f o0 = {}, o1 = {}, o2 = {}, o3 = {};
    float mrow[8], lrow[8];
    #pragma unroll
    for (int r = 0; r < 8; ++r) { mrow[r] = -1e30f; lrow[r] = 0.f; }

    int kmax = q0 + QB;                     // causal bound for this block
    int krow = tid >> 3, dcol = (tid & 7) * 8;
    bf16_t* Psw = Ps + w * 16 * PS_S;

    for (int kb0 = 0; kb0 < kmax; kb0 += KB) {
        // stage K block [KB][DH] (straight copy -> async-to-LDS when available)
        stage_b128(Km + (rowbase + kb0 + krow) * DIM + h * DH + dcol,
                   Ks + krow * KS_S + dcol);
        // stage V transposed: Vt[d][key]
        uint4 vv = *(const uint4*)(V + (rowbase + kb0 + krow) * DIM + h * DH + dcol);
        const bf16_t* vvh = (const bf16_t*)&vv;
        #pragma unroll
        for (int j = 0; j < 8; ++j) Vt[(dcol + j) * VT_S + krow] = vvh[j];
        async_stage_wait();
        __syncthreads();

        // ---- scores: S[16 q][32 keys] as two 16x16 tiles
        v8f s0 = {}, s1 = {};
        {
            union { uint4 u[2]; v16bf v; } kf;
            int ks = half * 16;
            // key tile 0 (cols 0..15)
            kf.u[0] = *(const uint4*)(Ks + l15 * KS_S + ks);
            kf.u[1] = *(const uint4*)(Ks + l15 * KS_S + ks + 8);
            s0 = __builtin_amdgcn_wmma_f32_16x16x32_bf16(false, qf0.v, false, kf.v,
                                                         (short)0, s0, false, false);
            kf.u[0] = *(const uint4*)(Ks + l15 * KS_S + 32 + ks);
            kf.u[1] = *(const uint4*)(Ks + l15 * KS_S + 32 + ks + 8);
            s0 = __builtin_amdgcn_wmma_f32_16x16x32_bf16(false, qf1.v, false, kf.v,
                                                         (short)0, s0, false, false);
            // key tile 1 (cols 16..31)
            kf.u[0] = *(const uint4*)(Ks + (16 + l15) * KS_S + ks);
            kf.u[1] = *(const uint4*)(Ks + (16 + l15) * KS_S + ks + 8);
            s1 = __builtin_amdgcn_wmma_f32_16x16x32_bf16(false, qf0.v, false, kf.v,
                                                         (short)0, s1, false, false);
            kf.u[0] = *(const uint4*)(Ks + (16 + l15) * KS_S + 32 + ks);
            kf.u[1] = *(const uint4*)(Ks + (16 + l15) * KS_S + 32 + ks + 8);
            s1 = __builtin_amdgcn_wmma_f32_16x16x32_bf16(false, qf1.v, false, kf.v,
                                                         (short)0, s1, false, false);
        }

        // ---- scale + causal mask + online softmax
        #pragma unroll
        for (int r = 0; r < 8; ++r) {
            int q = qw0 + r + 8 * half;
            int key0 = kb0 + l15;
            int key1 = kb0 + 16 + l15;
            float a = (key0 <= q) ? s0[r] * 0.125f : -1e30f;
            float b = (key1 <= q) ? s1[r] * 0.125f : -1e30f;
            float rm = fmaxf(a, b);
            #pragma unroll
            for (int off = 8; off > 0; off >>= 1) rm = fmaxf(rm, __shfl_xor(rm, off, 32));
            float mn = fmaxf(mrow[r], rm);
            float sf = __expf(mrow[r] - mn);
            float p0 = __expf(a - mn);
            float p1 = __expf(b - mn);
            float rs = p0 + p1;
            #pragma unroll
            for (int off = 8; off > 0; off >>= 1) rs += __shfl_xor(rs, off, 32);
            lrow[r] = lrow[r] * sf + rs;
            mrow[r] = mn;
            o0[r] *= sf; o1[r] *= sf; o2[r] *= sf; o3[r] *= sf;
            // park P (bf16) in per-wave LDS scratch in row/col order
            Psw[(r + 8 * half) * PS_S + l15]      = (bf16_t)p0;
            Psw[(r + 8 * half) * PS_S + 16 + l15] = (bf16_t)p1;
        }

        // ---- O += P(16x32) @ V(32x64)  (4 WMMAs over d tiles)
        union { uint4 u[2]; v16bf v; } pf, vf;
        {
            int kbq = half * 16;
            pf.u[0] = *(const uint4*)(Psw + l15 * PS_S + kbq);
            pf.u[1] = *(const uint4*)(Psw + l15 * PS_S + kbq + 8);
        }
        int ks = half * 16;
        vf.u[0] = *(const uint4*)(Vt + (0 * 16 + l15) * VT_S + ks);
        vf.u[1] = *(const uint4*)(Vt + (0 * 16 + l15) * VT_S + ks + 8);
        o0 = __builtin_amdgcn_wmma_f32_16x16x32_bf16(false, pf.v, false, vf.v,
                                                     (short)0, o0, false, false);
        vf.u[0] = *(const uint4*)(Vt + (1 * 16 + l15) * VT_S + ks);
        vf.u[1] = *(const uint4*)(Vt + (1 * 16 + l15) * VT_S + ks + 8);
        o1 = __builtin_amdgcn_wmma_f32_16x16x32_bf16(false, pf.v, false, vf.v,
                                                     (short)0, o1, false, false);
        vf.u[0] = *(const uint4*)(Vt + (2 * 16 + l15) * VT_S + ks);
        vf.u[1] = *(const uint4*)(Vt + (2 * 16 + l15) * VT_S + ks + 8);
        o2 = __builtin_amdgcn_wmma_f32_16x16x32_bf16(false, pf.v, false, vf.v,
                                                     (short)0, o2, false, false);
        vf.u[0] = *(const uint4*)(Vt + (3 * 16 + l15) * VT_S + ks);
        vf.u[1] = *(const uint4*)(Vt + (3 * 16 + l15) * VT_S + ks + 8);
        o3 = __builtin_amdgcn_wmma_f32_16x16x32_bf16(false, pf.v, false, vf.v,
                                                     (short)0, o3, false, false);
        __syncthreads();
    }

    #pragma unroll
    for (int r = 0; r < 8; ++r) {
        int q = qw0 + r + 8 * half;
        float invl = 1.0f / lrow[r];
        size_t ob = (rowbase + q) * DIM + h * DH;
        O[ob + l15]      = (bf16_t)(o0[r] * invl);
        O[ob + 16 + l15] = (bf16_t)(o1[r] * invl);
        O[ob + 32 + l15] = (bf16_t)(o2[r] * invl);
        O[ob + 48 + l15] = (bf16_t)(o3[r] * invl);
    }
}

// ---------------------------------------------------------------- launcher
extern "C" void kernel_launch(void* const* d_in, const int* in_sizes, int n_in,
                              void* d_out, int out_size, void* d_ws, size_t ws_size,
                              hipStream_t stream) {
    const float* x          = (const float*)d_in[0];
    const float* norm_scale = (const float*)d_in[1];
    const float* Wq         = (const float*)d_in[2];
    const float* Wk         = (const float*)d_in[3];
    const float* Wv         = (const float*)d_in[4];
    const float* Wo         = (const float*)d_in[5];
    float* out = (float*)d_out;

    char* ws = (char*)d_ws;
    size_t off = 0;
    auto alloc = [&](size_t bytes) -> char* {
        char* p = ws + off;
        off += (bytes + 255) & ~(size_t)255;
        return p;
    };
    bf16_t* xn   = (bf16_t*)alloc((size_t)ROWS * DIM * 2);
    bf16_t* Wqb  = (bf16_t*)alloc((size_t)DIM * DIM * 2);
    bf16_t* Wkb  = (bf16_t*)alloc((size_t)DIM * DIM * 2);
    bf16_t* Wvb  = (bf16_t*)alloc((size_t)DIM * DIM * 2);
    bf16_t* Wob  = (bf16_t*)alloc((size_t)DIM * DIM * 2);
    float*  Ftmp = (float*)alloc((size_t)ROWS * DIM * 4);  // serially reused fp32 temp
    bf16_t* Qr   = (bf16_t*)alloc((size_t)ROWS * DIM * 2);
    bf16_t* Kr   = (bf16_t*)alloc((size_t)ROWS * DIM * 2);
    bf16_t* Vb   = (bf16_t*)alloc((size_t)ROWS * DIM * 2);
    bf16_t* attn = (bf16_t*)alloc((size_t)ROWS * DIM * 2);

    const int wn = DIM * DIM;
    f32_to_bf16_kernel<<<wn / 1024, 256, 0, stream>>>(Wq, Wqb, wn);
    f32_to_bf16_kernel<<<wn / 1024, 256, 0, stream>>>(Wk, Wkb, wn);
    f32_to_bf16_kernel<<<wn / 1024, 256, 0, stream>>>(Wv, Wvb, wn);
    f32_to_bf16_kernel<<<wn / 1024, 256, 0, stream>>>(Wo, Wob, wn);

    rmsnorm_kernel<<<ROWS, 256, 0, stream>>>(x, norm_scale, xn);

    dim3 gg(DIM / BN, ROWS / BM);
    const int rope_blocks = (ROWS * DIM / 2) / 256;

    gemm_bf16_kernel<<<gg, 256, 0, stream>>>(xn, Wqb, Ftmp, ROWS, DIM, DIM);
    rope_kernel<<<rope_blocks, 256, 0, stream>>>(Ftmp, Qr);
    gemm_bf16_kernel<<<gg, 256, 0, stream>>>(xn, Wkb, Ftmp, ROWS, DIM, DIM);
    rope_kernel<<<rope_blocks, 256, 0, stream>>>(Ftmp, Kr);
    gemm_bf16_kernel<<<gg, 256, 0, stream>>>(xn, Wvb, Ftmp, ROWS, DIM, DIM);
    f32_to_bf16_kernel<<<(ROWS * DIM) / 1024, 256, 0, stream>>>(Ftmp, Vb, ROWS * DIM);

    flash_attn_kernel<<<dim3(SEQ / QB, NB * NH), 256, 0, stream>>>(Qr, Kr, Vb, attn);

    gemm_bf16_kernel<<<gg, 256, 0, stream>>>(attn, Wob, out, ROWS, DIM, DIM);
}
